// AttentionLayer_31181462568969
// MI455X (gfx1250) — matrix operand order repaired and checked by reading, not compiled, not run on hardware
//
#include <hip/hip_runtime.h>
#include <hip/hip_bf16.h>

typedef __bf16 bf16;
typedef __attribute__((ext_vector_type(16))) __bf16 v16bf;
typedef __attribute__((ext_vector_type(8)))  float  v8f;

#define WMMA_BF16(a, b, c) \
  __builtin_amdgcn_wmma_f32_16x16x32_bf16(false, (a), false, (b), (short)0, (c), false, false)

// ---------------------------------------------------------------------------
// Fragment loaders (wave32; contiguous k order, same for A and B, not the ISA table order)
// A (16x32 bf16): lane m = lane&15, half h = lane>>4,
//   element e: K = e + 16h
// B (32x16 bf16) loaded from B^T row-major: lane n = lane&15,
//   element e: K = e + 16h  (contiguous 16 elems -> 2x b128 per lane)
// C/D (16x16 f32): lane n = lane&15, vgpr r -> M = r + 8h
// ---------------------------------------------------------------------------

static __device__ inline v16bf load_a_bf16(const bf16* base, int ld, int row0, int k0) {
  int lane = threadIdx.x & 31;
  int m = lane & 15, h = lane >> 4;
  const bf16* p = base + (size_t)(row0 + m) * ld + k0 + 16 * h;
  v16bf a;
#pragma unroll
  for (int e = 0; e < 16; ++e) a[e] = p[e];
  return a;
}

static __device__ inline v16bf load_b_bt_bf16(const bf16* bt, int ld, int col0, int k0) {
  int lane = threadIdx.x & 31;
  int n = lane & 15, h = lane >> 4;
  const bf16* p = bt + (size_t)(col0 + n) * ld + k0 + 16 * h;
  v16bf b;
#pragma unroll
  for (int e = 0; e < 16; ++e) b[e] = p[e];
  return b;
}

// ---------------------------------------------------------------------------
// Kernel 0: one-time data prep (runs once over ~48 MB, ~2us at 23.3 TB/s):
//  - x (B*S,512) f32 -> xb bf16 (same layout)
//  - Wq/Wk/Wv (512,64) f32 -> transposed bf16 (64,512)  [B^T layout]
//  - Wo (64,512) f32 -> transposed bf16 (512,64)        [B^T layout]
// ---------------------------------------------------------------------------
__global__ __launch_bounds__(256) void prep_kernel(
    const float* __restrict__ x,
    const float* __restrict__ Wq, const float* __restrict__ Wk,
    const float* __restrict__ Wv, const float* __restrict__ Wo,
    bf16* __restrict__ xb, bf16* __restrict__ Wqt, bf16* __restrict__ Wkt,
    bf16* __restrict__ Wvt, bf16* __restrict__ Wot, int totalx) {
  int i = blockIdx.x * blockDim.x + threadIdx.x;
  for (int idx = i; idx < totalx; idx += gridDim.x * blockDim.x)
    xb[idx] = (bf16)x[idx];
  if (i < 64 * 512) {
    int nH = i >> 9, d = i & 511;   // Wqt[nH][d] = Wq[d][nH]
    Wqt[i] = (bf16)Wq[d * 64 + nH];
    Wkt[i] = (bf16)Wk[d * 64 + nH];
    Wvt[i] = (bf16)Wv[d * 64 + nH];
    int nd = i >> 6, hh = i & 63;   // Wot[nd][hh] = Wo[hh][nd]
    Wot[i] = (bf16)Wo[hh * 512 + nd];
  }
}

// ---------------------------------------------------------------------------
// Kernel 1: fused QKV projection.  xb:(B*S,512) bf16 -> Q,K:(B*S,64) bf16,
// Vt:(B,64,S) bf16 (transposed so PV's B-operand is contiguous later).
// The 16-row x tile (16 KB, contiguous) is shared by all 4 waves of the
// block: staged once via async global->LDS DMA (ASYNCcnt), A-frags from LDS.
// ---------------------------------------------------------------------------
__global__ __launch_bounds__(128) void qkv_kernel(
    const bf16* __restrict__ xb,
    const bf16* __restrict__ Wqt, const float* __restrict__ bq,
    const bf16* __restrict__ Wkt, const float* __restrict__ bk,
    const bf16* __restrict__ Wvt, const float* __restrict__ bv,
    bf16* __restrict__ Q, bf16* __restrict__ K, bf16* __restrict__ Vt) {
  const int S = 4096, D = 512, H = 64;
  __shared__ bf16 xtile[16 * 512];                 // 16 KB
  int wave = threadIdx.x >> 5;
  int row0 = blockIdx.x * 16;                      // over B*S/16 = 1024 tiles
  int col0 = wave * 16;                            // 4 waves cover H = 64

  // Async-stage the contiguous 16 KB x tile: 8 iters x 128 threads x 16 B.
  {
    const char* src = (const char*)(xb + (size_t)row0 * D);
    unsigned lbase = (unsigned)(uintptr_t)(void*)xtile;  // low 32 bits = LDS offset
    int tid = threadIdx.x;
#pragma unroll
    for (int it = 0; it < 8; ++it) {
      unsigned loff = lbase + (unsigned)(it * 2048 + tid * 16);
      unsigned long long gaddr =
          (unsigned long long)(uintptr_t)(src + it * 2048 + tid * 16);
      asm volatile("global_load_async_to_lds_b128 %0, %1, off"
                   :: "v"(loff), "v"(gaddr) : "memory");
    }
    asm volatile("s_wait_asynccnt 0" ::: "memory");
    __syncthreads();
  }

  v8f cq = {}, ck = {}, cv = {};
  for (int d0 = 0; d0 < D; d0 += 32) {
    v16bf xa  = load_a_bf16(xtile, D, 0, d0);           // ds_load from LDS
    v16bf wqf = load_b_bt_bf16(Wqt, D, col0, d0);       // contiguous b128 loads
    v16bf wkf = load_b_bt_bf16(Wkt, D, col0, d0);
    v16bf wvf = load_b_bt_bf16(Wvt, D, col0, d0);
    cq = WMMA_BF16(xa, wqf, cq);
    ck = WMMA_BF16(xa, wkf, ck);
    cv = WMMA_BF16(xa, wvf, cv);
  }

  int lane = threadIdx.x & 31;
  int n = lane & 15, h = lane >> 4;
  float biasq = bq[col0 + n], biask = bk[col0 + n], biasv = bv[col0 + n];
  int bidx = row0 >> 12;          // row0 / S  (S = 4096)
  int s    = row0 & (S - 1);
#pragma unroll
  for (int r = 0; r < 8; ++r) {
    int m = r + 8 * h;
    Q[(size_t)(row0 + m) * H + col0 + n] = (bf16)(cq[r] + biasq);
    K[(size_t)(row0 + m) * H + col0 + n] = (bf16)(ck[r] + biask);
    Vt[(size_t)(bidx * H + col0 + n) * S + (s + m)] = (bf16)(cv[r] + biasv);
  }
}

// ---------------------------------------------------------------------------
// Kernel 2: causal flash attention.  One wave = one 16-row query tile.
// Inner loop over 32-key blocks: 4 WMMAs for scores (K-dim = H = 64),
// online softmax in f32 (scale 1/sqrt(S)=1/64, mask add -1e9 per reference),
// P staged via wave-private LDS (s_wait_dscnt), 4 WMMAs for P@V.
// ---------------------------------------------------------------------------
__global__ __launch_bounds__(128) void attn_kernel(
    const bf16* __restrict__ Q, const bf16* __restrict__ K,
    const bf16* __restrict__ Vt, bf16* __restrict__ O) {
  const int S = 4096, H = 64;
  const float NEG = -1.0e9f, SCALE = 0.015625f;  // 1/sqrt(4096)
  __shared__ bf16 Pbuf[4][16 * 32];

  int wave = threadIdx.x >> 5;
  int lane = threadIdx.x & 31;
  int n = lane & 15, h = lane >> 4;
  int bx = blockIdx.x;                 // 256 blocks: 4 batches x 64
  int b  = bx >> 6;
  int q0 = ((bx & 63) * 4 + wave) * 16;

  const bf16* Qb  = Q  + (size_t)b * S * H;
  const bf16* Kb  = K  + (size_t)b * S * H;
  const bf16* Vtb = Vt + (size_t)b * H * S;
  bf16*       Ob  = O  + (size_t)b * S * H;
  bf16*       P   = Pbuf[wave];

  v16bf qa0 = load_a_bf16(Qb, H, q0, 0);
  v16bf qa1 = load_a_bf16(Qb, H, q0, 32);

  float m_i[8], l_i[8];
#pragma unroll
  for (int r = 0; r < 8; ++r) { m_i[r] = -3.0e38f; l_i[r] = 0.0f; }
  v8f o0 = {}, o1 = {}, o2 = {}, o3 = {};

  int nblk = ((q0 + 15) >> 5) + 1;     // causal: key blocks with t0 <= q0+15
  for (int kb = 0; kb < nblk; ++kb) {
    int t0 = kb * 32;
    if (t0 + 32 < S)                   // prefetch next K block (global_prefetch)
      __builtin_prefetch(Kb + (size_t)(t0 + 32 + n) * H, 0, 1);

    // scores: two 16x16 subtiles over keys [t0, t0+32)
    v8f s0 = {}, s1 = {};
    s0 = WMMA_BF16(qa0, load_b_bt_bf16(Kb, H, t0, 0), s0);
    s0 = WMMA_BF16(qa1, load_b_bt_bf16(Kb, H, t0, 32), s0);
    s1 = WMMA_BF16(qa0, load_b_bt_bf16(Kb, H, t0 + 16, 0), s1);
    s1 = WMMA_BF16(qa1, load_b_bt_bf16(Kb, H, t0 + 16, 32), s1);

    // scale + causal mask (element (m,n): q = q0+r+8h, t = t0[+16]+n)
    float sv0[8], sv1[8];
#pragma unroll
    for (int r = 0; r < 8; ++r) {
      int q = q0 + r + 8 * h;
      sv0[r] = s0[r] * SCALE + ((t0 + n      > q) ? NEG : 0.0f);
      sv1[r] = s1[r] * SCALE + ((t0 + 16 + n > q) ? NEG : 0.0f);
    }

    // row max (reduce over n: 16 lanes within each half-wave)
    float mnew[8], alpha[8];
#pragma unroll
    for (int r = 0; r < 8; ++r) {
      float mx = fmaxf(sv0[r], sv1[r]);
      mx = fmaxf(mx, __shfl_xor(mx, 1));
      mx = fmaxf(mx, __shfl_xor(mx, 2));
      mx = fmaxf(mx, __shfl_xor(mx, 4));
      mx = fmaxf(mx, __shfl_xor(mx, 8));
      mnew[r]  = fmaxf(m_i[r], mx);
      alpha[r] = __expf(m_i[r] - mnew[r]);
      m_i[r]   = mnew[r];
    }

    // exp, row sum, stage P (bf16) into wave-private LDS
#pragma unroll
    for (int r = 0; r < 8; ++r) {
      float p0 = __expf(sv0[r] - mnew[r]);
      float p1 = __expf(sv1[r] - mnew[r]);
      float rs = p0 + p1;
      rs += __shfl_xor(rs, 1);
      rs += __shfl_xor(rs, 2);
      rs += __shfl_xor(rs, 4);
      rs += __shfl_xor(rs, 8);
      l_i[r] = l_i[r] * alpha[r] + rs;
      int m = r + 8 * h;
      P[m * 32 + n]      = (bf16)p0;
      P[m * 32 + 16 + n] = (bf16)p1;
    }
    asm volatile("s_wait_dscnt 0" ::: "memory");  // LDS stores -> A-frag reload

    // rescale running O by alpha, then accumulate P @ V (B-op from Vt, contiguous)
#pragma unroll
    for (int r = 0; r < 8; ++r) {
      o0[r] *= alpha[r]; o1[r] *= alpha[r]; o2[r] *= alpha[r]; o3[r] *= alpha[r];
    }
    v16bf pa = load_a_bf16(P, 32, 0, 0);
    o0 = WMMA_BF16(pa, load_b_bt_bf16(Vtb, S, 0,  t0), o0);
    o1 = WMMA_BF16(pa, load_b_bt_bf16(Vtb, S, 16, t0), o1);
    o2 = WMMA_BF16(pa, load_b_bt_bf16(Vtb, S, 32, t0), o2);
    o3 = WMMA_BF16(pa, load_b_bt_bf16(Vtb, S, 48, t0), o3);
  }

  // normalize and store O (bf16) for the output projection
#pragma unroll
  for (int r = 0; r < 8; ++r) {
    float inv = 1.0f / l_i[r];
    int m = r + 8 * h;
    Ob[(size_t)(q0 + m) * H + n]      = (bf16)(o0[r] * inv);
    Ob[(size_t)(q0 + m) * H + 16 + n] = (bf16)(o1[r] * inv);
    Ob[(size_t)(q0 + m) * H + 32 + n] = (bf16)(o2[r] * inv);
    Ob[(size_t)(q0 + m) * H + 48 + n] = (bf16)(o3[r] * inv);
  }
}

// ---------------------------------------------------------------------------
// Kernel 3: output projection.  O:(B*S,64) bf16 @ Wo^T(bf16) + bo -> f32.
// One wave = one 16x16 output tile, K-dim = 64 -> 2 WMMAs.
// ---------------------------------------------------------------------------
__global__ __launch_bounds__(128) void oproj_kernel(
    const bf16* __restrict__ O, const bf16* __restrict__ Wot,
    const float* __restrict__ bo, float* __restrict__ out) {
  const int D = 512, H = 64;
  int wave = threadIdx.x >> 5;
  int lane = threadIdx.x & 31;
  int n = lane & 15, h = lane >> 4;
  int bx = blockIdx.x;                // 1024 row tiles x 8 blocks
  int row0 = (bx >> 3) * 16;
  int col0 = ((bx & 7) * 4 + wave) * 16;

  v8f c = {};
  c = WMMA_BF16(load_a_bf16(O, H, row0, 0),  load_b_bt_bf16(Wot, H, col0, 0),  c);
  c = WMMA_BF16(load_a_bf16(O, H, row0, 32), load_b_bt_bf16(Wot, H, col0, 32), c);

  float bias = bo[col0 + n];
#pragma unroll
  for (int r = 0; r < 8; ++r) {
    int m = r + 8 * h;
    out[(size_t)(row0 + m) * D + col0 + n] = c[r] + bias;
  }
}

// ---------------------------------------------------------------------------
extern "C" void kernel_launch(void* const* d_in, const int* in_sizes, int n_in,
                              void* d_out, int out_size, void* d_ws, size_t ws_size,
                              hipStream_t stream) {
  const float* x  = (const float*)d_in[0];
  const float* Wq = (const float*)d_in[1];
  const float* bq = (const float*)d_in[2];
  const float* Wk = (const float*)d_in[3];
  const float* bk = (const float*)d_in[4];
  const float* Wv = (const float*)d_in[5];
  const float* bv = (const float*)d_in[6];
  const float* Wo = (const float*)d_in[7];
  const float* bo = (const float*)d_in[8];
  float* out = (float*)d_out;

  const size_t BS = 4 * 4096;   // B*S rows
  const size_t H = 64, D = 512;
  // Workspace (bf16): xb 16 MB; Wq/Wk/Wv/Wo^T 64 KB each; Q,K,Vt,O 2 MB each.
  bf16* xb  = (bf16*)d_ws;
  bf16* Wqt = xb  + BS * D;
  bf16* Wkt = Wqt + H * D;
  bf16* Wvt = Wkt + H * D;
  bf16* Wot = Wvt + H * D;
  bf16* Q   = Wot + D * H;
  bf16* K   = Q   + BS * H;
  bf16* Vt  = K   + BS * H;
  bf16* O   = Vt  + BS * H;

  prep_kernel<<<4096, 256, 0, stream>>>(x, Wq, Wk, Wv, Wo,
                                        xb, Wqt, Wkt, Wvt, Wot, (int)(BS * D));
  qkv_kernel<<<BS / 16, 128, 0, stream>>>(xb, Wqt, bq, Wkt, bk, Wvt, bv, Q, K, Vt);
  attn_kernel<<<256, 128, 0, stream>>>(Q, K, Vt, O);
  oproj_kernel<<<8192, 128, 0, stream>>>(O, Wot, bo, out);
}
